// Classifier_6863357739230
// MI455X (gfx1250) — compile-verified, matrix operand order checked by reading, not run on hardware
//
#include <hip/hip_runtime.h>
#include <math.h>

typedef __attribute__((ext_vector_type(16))) _Float16     v16h;
typedef __attribute__((ext_vector_type(8)))  float        v8f;
typedef __attribute__((ext_vector_type(4)))  unsigned int v4u;
typedef __attribute__((ext_vector_type(4)))  int          v4i;
typedef __attribute__((ext_vector_type(8)))  int          v8i;

#define B_ROWS 8192
#define D_DIM  768
#define K_ANCH 256
#define S_SUB  16
#define N_COLS (K_ANCH * S_SUB)   // 4096

#define MT 128
#define NT 128
#define KT 32

// ---------------------------------------------------------------------------
// TDM descriptor helpers (CDNA5 ISA §8.3 / §8.4).
// Group 0 (128b): [1:0]=count=1 | [63:32]=lds_addr | [120:64]=global_addr |
//                 [127:126]=type=2
// Group 1 (256b): [17:16]=data_size(1 -> 2 bytes) | [79:48]=tensor_dim0 |
//                 [111:80]=tensor_dim1 | [127:112]=tile_dim0 |
//                 [143:128]=tile_dim1 | [207:160]=tensor_dim0_stride
// ---------------------------------------------------------------------------
__device__ __forceinline__ v4u tdm_g0(unsigned lds_addr, unsigned long long ga) {
    v4u g;
    g.x = 1u;                                               // count=1, user D#
    g.y = lds_addr;
    g.z = (unsigned)(ga & 0xffffffffull);
    g.w = (unsigned)((ga >> 32) & 0x01ffffffull) | (2u << 30);  // type=2
    return g;
}

__device__ __forceinline__ v8i tdm_g1(unsigned tdim0, unsigned tdim1,
                                      unsigned tile0, unsigned tile1,
                                      unsigned long long stride0) {
    unsigned w0 = 1u << 16;                                  // data_size=2B
    unsigned w1 = (tdim0 & 0xffffu) << 16;                   // tensor_dim0 lo
    unsigned w2 = ((tdim0 >> 16) & 0xffffu) | ((tdim1 & 0xffffu) << 16);
    unsigned w3 = ((tdim1 >> 16) & 0xffffu) | ((tile0 & 0xffffu) << 16);
    unsigned w4 = (tile1 & 0xffffu);                         // tile_dim2 = 0
    unsigned w5 = (unsigned)(stride0 & 0xffffffffull);
    unsigned w6 = (unsigned)((stride0 >> 32) & 0xffffull);   // dim1_stride=0
    v8i g;
    g[0] = (int)w0; g[1] = (int)w1; g[2] = (int)w2; g[3] = (int)w3;
    g[4] = (int)w4; g[5] = (int)w5; g[6] = (int)w6; g[7] = 0;
    return g;
}

#if __has_include(<hip/amd_detail/amd_gfx1250_TDM.h>)
// amdgpu-toolchain (clang-23 / therock-10.0): 6-arg builtin
#define TDM_LOAD(g0, g1, z4, z8) \
    __builtin_amdgcn_tensor_load_to_lds((g0), (g1), (z4), (z4), (z8), 0)
#else
// ROCm 7.2 (clang-22): 5-arg builtin
#define TDM_LOAD(g0, g1, z4, z8) \
    __builtin_amdgcn_tensor_load_to_lds((g0), (g1), (z4), (z4), 0)
#endif

// ---------------------------------------------------------------------------
// Kernel 1: row L2-normalize f32 -> f16.  One 256-thread block per row,
// 768 = 256 * 3 elements per row.
// ---------------------------------------------------------------------------
__global__ __launch_bounds__(256) void normalize_rows_f16(
    const float* __restrict__ in, _Float16* __restrict__ out)
{
    const int row = blockIdx.x;
    const float* src = in  + (size_t)row * D_DIM;
    _Float16*    dst = out + (size_t)row * D_DIM;

    float v[3];
    float partial = 0.0f;
#pragma unroll
    for (int i = 0; i < 3; ++i) {
        v[i] = src[threadIdx.x + i * 256];
        partial += v[i] * v[i];
    }
#pragma unroll
    for (int off = 16; off > 0; off >>= 1)
        partial += __shfl_xor(partial, off, 32);

    __shared__ float wsum[8];
    const int wave = threadIdx.x >> 5;
    const int lane = threadIdx.x & 31;
    if (lane == 0) wsum[wave] = partial;
    __syncthreads();

    float tot = 0.0f;
#pragma unroll
    for (int w = 0; w < 8; ++w) tot += wsum[w];

    const float inv = 1.0f / fmaxf(sqrtf(tot), 1e-8f);   // COS_EPS clamp
#pragma unroll
    for (int i = 0; i < 3; ++i)
        dst[threadIdx.x + i * 256] = (_Float16)(v[i] * inv);
}

// ---------------------------------------------------------------------------
// Kernel 2: cos GEMM, TDM-staged + double buffered, v_wmma_f32_16x16x32_f16.
// Grid: (B_ROWS/MT) x (N_COLS/NT). Block: 256 threads = 8 waves in 4(M)x2(N).
// Wave 0 issues tensor_load_to_lds for both tiles; TENSORcnt gates reuse.
// ---------------------------------------------------------------------------
__global__ __launch_bounds__(256) void cos_gemm_f16(
    const _Float16* __restrict__ A,   // [B_ROWS][D_DIM] normalized emb
    const _Float16* __restrict__ Bm,  // [N_COLS][D_DIM] normalized anchors
    _Float16* __restrict__ Sc)        // [B_ROWS][N_COLS] scores
{
    __shared__ alignas(64) _Float16 As[2][MT][KT];
    __shared__ alignas(64) _Float16 Bs[2][NT][KT];

    const int tid  = threadIdx.x;
    const int wave = tid >> 5;
    const int lane = tid & 31;
    const int wm   = wave & 3;        // 4 waves along M
    const int wn   = wave >> 2;       // 2 waves along N
    const int m0   = blockIdx.x * MT;
    const int n0   = blockIdx.y * NT;

    v8f acc[2][4];
#pragma unroll
    for (int mi = 0; mi < 2; ++mi)
#pragma unroll
        for (int ni = 0; ni < 4; ++ni)
            acc[mi][ni] = (v8f){0.f,0.f,0.f,0.f,0.f,0.f,0.f,0.f};

    // TDM descriptors (uniform, block-level)
    const unsigned long long gaA =
        (unsigned long long)A  + (unsigned long long)m0 * D_DIM * 2ull;
    const unsigned long long gaB =
        (unsigned long long)Bm + (unsigned long long)n0 * D_DIM * 2ull;
    const unsigned ldsA[2] = { (unsigned)(unsigned long long)&As[0][0][0],
                               (unsigned)(unsigned long long)&As[1][0][0] };
    const unsigned ldsB[2] = { (unsigned)(unsigned long long)&Bs[0][0][0],
                               (unsigned)(unsigned long long)&Bs[1][0][0] };
    const v8i g1A = tdm_g1(D_DIM, B_ROWS, KT, MT, D_DIM);
    const v8i g1B = tdm_g1(D_DIM, N_COLS, KT, NT, D_DIM);
    const v4i z4 = {0, 0, 0, 0};
    const v8i z8 = {0, 0, 0, 0, 0, 0, 0, 0};

    auto issue = [&](int buf, int kk) {
        if (tid == 0) {
            TDM_LOAD(tdm_g0(ldsA[buf], gaA + (unsigned long long)kk * 2ull),
                     g1A, z4, z8);
            TDM_LOAD(tdm_g0(ldsB[buf], gaB + (unsigned long long)kk * 2ull),
                     g1B, z4, z8);
        }
    };

    auto compute = [&](int buf) {
        const int hr = lane & 15;
        const int hs = (lane >> 4) * 16;
        v16h af[2], bf[4];
#pragma unroll
        for (int mi = 0; mi < 2; ++mi)
            af[mi] = *(const v16h*)&As[buf][wm * 32 + mi * 16 + hr][hs];
#pragma unroll
        for (int ni = 0; ni < 4; ++ni)
            bf[ni] = *(const v16h*)&Bs[buf][wn * 64 + ni * 16 + hr][hs];
#pragma unroll
        for (int mi = 0; mi < 2; ++mi)
#pragma unroll
            for (int ni = 0; ni < 4; ++ni)
                acc[mi][ni] = __builtin_amdgcn_wmma_f32_16x16x32_f16(
                    false, af[mi], false, bf[ni],
                    (short)0, acc[mi][ni], false, false);
    };

    int buf = 0;
    issue(0, 0);                                  // prologue: tile 0 in flight
    for (int kk = 0; kk < D_DIM - KT; kk += KT) {
        issue(buf ^ 1, kk + KT);                  // overlap next tile fetch
        __builtin_amdgcn_s_wait_tensorcnt(2);     // current tile (in-order) done
        __syncthreads();
        compute(buf);
        __syncthreads();                          // all reads done before reuse
        buf ^= 1;
    }
    __builtin_amdgcn_s_wait_tensorcnt(0);         // last tile done
    __syncthreads();
    compute(buf);

    // Epilogue: C layout -> lane (l&15)=N, (l>>4)*8+e = M (ISA 7.12.2)
#pragma unroll
    for (int mi = 0; mi < 2; ++mi) {
#pragma unroll
        for (int ni = 0; ni < 4; ++ni) {
#pragma unroll
            for (int e = 0; e < 8; ++e) {
                const int row = m0 + wm * 32 + mi * 16 + (lane >> 4) * 8 + e;
                const int col = n0 + wn * 64 + ni * 16 + (lane & 15);
                const float c = acc[mi][ni][e];
                Sc[(size_t)row * N_COLS + col] =
                    (_Float16)((1.0f + c) * 0.5f + 1e-8f);
            }
        }
    }
}

// ---------------------------------------------------------------------------
// Kernel 3: entropy pooling.  One 256-thread block per batch row b.
// Thread k owns the 16 contiguous scores Sc[b][k*16 + s], s=0..15.
// ---------------------------------------------------------------------------
__global__ __launch_bounds__(256) void entropy_pool(
    const _Float16* __restrict__ Sc, float* __restrict__ out)
{
    const int b    = blockIdx.x;
    const int k    = threadIdx.x;       // 0..255 == K_ANCH
    const int wave = k >> 5;
    const int lane = k & 31;

    __shared__ float red[8][S_SUB];
    __shared__ float smax[S_SUB], ssum[S_SUB], sent[S_SUB];

    float lg[S_SUB];
    const _Float16* p = Sc + (size_t)b * N_COLS + k * S_SUB;
#pragma unroll
    for (int s = 0; s < S_SUB; ++s)
        lg[s] = (float)p[s] * 2.0f;          // /TEMP, TEMP=0.5

    // per-s max over k
#pragma unroll
    for (int s = 0; s < S_SUB; ++s) {
        float v = lg[s];
#pragma unroll
        for (int off = 16; off > 0; off >>= 1)
            v = fmaxf(v, __shfl_xor(v, off, 32));
        if (lane == 0) red[wave][s] = v;
    }
    __syncthreads();
    if (k < S_SUB) {
        float m = red[0][k];
#pragma unroll
        for (int w = 1; w < 8; ++w) m = fmaxf(m, red[w][k]);
        smax[k] = m;
    }
    __syncthreads();

    // per-s sum of exp
    float ex[S_SUB];
#pragma unroll
    for (int s = 0; s < S_SUB; ++s) {
        ex[s] = __expf(lg[s] - smax[s]);
        float v = ex[s];
#pragma unroll
        for (int off = 16; off > 0; off >>= 1)
            v += __shfl_xor(v, off, 32);
        if (lane == 0) red[wave][s] = v;
    }
    __syncthreads();
    if (k < S_SUB) {
        float t = 0.0f;
#pragma unroll
        for (int w = 0; w < 8; ++w) t += red[w][k];
        ssum[k] = t;
    }
    __syncthreads();

    // probs + per-s entropy
    float pr[S_SUB];
#pragma unroll
    for (int s = 0; s < S_SUB; ++s) {
        pr[s] = ex[s] / ssum[s];
        float v = -pr[s] * logf(pr[s] + 1e-8f);
#pragma unroll
        for (int off = 16; off > 0; off >>= 1)
            v += __shfl_xor(v, off, 32);
        if (lane == 0) red[wave][s] = v;
    }
    __syncthreads();
    if (k < S_SUB) {
        float t = 0.0f;
#pragma unroll
        for (int w = 0; w < 8; ++w) t += red[w][k];
        sent[k] = t;
    }
    __syncthreads();

    // entropy weights + fused log-probs
    float w[S_SUB], wsum = 0.0f;
#pragma unroll
    for (int s = 0; s < S_SUB; ++s) {
        w[s] = 1.0f / (sent[s] + 1e-6f);     // DOMAIN_ENTROPY_EPS
        wsum += w[s];
    }
    const float inv = 1.0f / (wsum + 1e-8f);
    float fused = 0.0f;
#pragma unroll
    for (int s = 0; s < S_SUB; ++s)
        fused += (w[s] * inv) * pr[s];

    out[(size_t)b * K_ANCH + k] = logf(fused + 1e-8f);
}

// ---------------------------------------------------------------------------
extern "C" void kernel_launch(void* const* d_in, const int* in_sizes, int n_in,
                              void* d_out, int out_size, void* d_ws, size_t ws_size,
                              hipStream_t stream) {
    (void)in_sizes; (void)n_in; (void)out_size; (void)ws_size;

    const float* emb     = (const float*)d_in[0];   // [8192, 768]
    const float* anchors = (const float*)d_in[1];   // [256, 16, 768] == [4096, 768]
    float* out = (float*)d_out;                     // [8192, 256]

    _Float16* eh = (_Float16*)d_ws;                          // [8192*768]
    _Float16* ah = eh + (size_t)B_ROWS * D_DIM;              // [4096*768]
    _Float16* sc = ah + (size_t)N_COLS * D_DIM;              // [8192*4096]

    normalize_rows_f16<<<B_ROWS, 256, 0, stream>>>(emb, eh);
    normalize_rows_f16<<<N_COLS, 256, 0, stream>>>(anchors, ah);

    dim3 gemm_grid(B_ROWS / MT, N_COLS / NT);                // 64 x 32
    cos_gemm_f16<<<gemm_grid, 256, 0, stream>>>(eh, ah, sc);

    entropy_pool<<<B_ROWS, 256, 0, stream>>>(sc, out);
}
